// StandardROIHeads_83408264888699
// MI455X (gfx1250) — compile-verified, matrix operand order checked by reading, not run on hardware
//
#include <hip/hip_runtime.h>
#include <stdint.h>
#include <stddef.h>

#define B_      4
#define N_      256
#define C_      18
#define CP1     19
#define M_      (N_ * C_)      /* 4608 flattened candidates */
#define MW_     (M_ / 32)      /* 144 mask words per row */
#define TOPK_   256
#define SORTN   8192
#define SCORE_TH 0.05f
#define NMS_TH   0.25f

typedef _Float16 v16h __attribute__((ext_vector_type(16)));
typedef float    v8f  __attribute__((ext_vector_type(8)));

// ---------------- workspace layout (bytes) ----------------
static constexpr size_t OFF_SCORES = 0;
static constexpr size_t OFF_BOXES  = OFF_SCORES + (size_t)B_ * M_ * sizeof(float);
static constexpr size_t OFF_CLS    = OFF_BOXES  + (size_t)B_ * M_ * 6 * sizeof(float);
static constexpr size_t OFF_KEYS   = OFF_CLS    + (size_t)B_ * M_ * sizeof(int);
static constexpr size_t OFF_SREC   = OFF_KEYS   + (size_t)B_ * M_ * sizeof(unsigned);
static constexpr size_t OFF_VBITS  = OFF_SREC   + (size_t)B_ * M_ * 8 * sizeof(float);
static constexpr size_t OFF_MASK   = OFF_VBITS  + (size_t)B_ * MW_ * sizeof(unsigned);

// =====================================================================
// K1: softmax (denominator via WMMA row-reduction) + box decode + keys
// grid = B_ * 2 blocks of 256 (8 waves x 16 rows = 128 proposals/block)
// =====================================================================
__global__ void __launch_bounds__(256)
nms_decode_kernel(const float* __restrict__ logits, const float* __restrict__ deltas,
                  const float* __restrict__ props,  const float* __restrict__ origs,
                  float* __restrict__ scores, float* __restrict__ boxes,
                  int* __restrict__ cls, unsigned* __restrict__ keys)
{
    const int wave = threadIdx.x >> 5;
    const int lane = threadIdx.x & 31;
    const int b    = blockIdx.x >> 1;
    const int chnk = blockIdx.x & 1;
    const int row  = lane & 15;
    const bool hi  = lane >= 16;
    const int n    = chnk * 128 + wave * 16 + row;   // proposal id in [0,256)

    const float* lrow = logits + ((size_t)b * N_ + n) * CP1;

    // lane<16 owns K=0..7 and K=16..18 of its row; lane>=16 owns K=8..15
    int cols[11];
    int cnt;
    if (!hi) {
        cnt = 11;
        for (int t = 0; t < 8; ++t) cols[t] = t;
        cols[8] = 16; cols[9] = 17; cols[10] = 18;
    } else {
        cnt = 8;
        for (int t = 0; t < 8; ++t) cols[t] = 8 + t;
    }

    float lv[11];
    float mx = -1e30f;
    for (int t = 0; t < cnt; ++t) { lv[t] = lrow[cols[t]]; mx = fmaxf(mx, lv[t]); }
    mx = fmaxf(mx, __shfl_xor(mx, 16));   // combine the two half-rows (wave32)

    // A tile: 16x32 f16, exp(logit - rowmax) in valid K slots, zero padding.
    v16h a;
    for (int t = 0; t < 16; ++t) a[t] = (_Float16)0.0f;
    for (int t = 0; t < cnt; ++t) a[t] = (_Float16)__expf(lv[t] - mx);

    v16h bn;
    for (int t = 0; t < 16; ++t) bn[t] = (_Float16)1.0f;

    v8f c = {0.f, 0.f, 0.f, 0.f, 0.f, 0.f, 0.f, 0.f};
    // D[m][n] = sum_k A[m,k] -> every D element of column n holds rowsum(m)
    c = __builtin_amdgcn_wmma_f32_16x16x32_f16(false, a, false, bn,
                                               (short)0, c, false, false);

    __shared__ float denom[8][16];
    if (lane == 0)  { for (int r = 0; r < 8; ++r) denom[wave][r]     = c[r]; }
    if (lane == 16) { for (int r = 0; r < 8; ++r) denom[wave][8 + r] = c[r]; }
    __syncthreads();
    const float d = denom[wave][row];

    const float* pp = props + ((size_t)b * N_ + n) * 6;
    const float* op = origs + ((size_t)b * N_ + n) * 3;
    const float p0 = pp[0], p1 = pp[1], p2 = pp[2], p3 = pp[3], p4 = pp[4], p5 = pp[5];
    const float o0 = op[0], o1 = op[1], o2 = op[2];

    for (int t = 0; t < cnt; ++t) {
        const int col = cols[t];
        if (col >= C_) continue;                      // drop background class
        const float s = __expf(lv[t] - mx) / d;
        const int m = n * C_ + col;
        const size_t mb = (size_t)b * M_ + m;
        scores[mb] = s;
        cls[mb]    = col;
        const float* dp = deltas + (((size_t)b * N_ + n) * C_ + col) * 6;
        float* bx = boxes + mb * 6;
        bx[0] = o0 - (p0 + dp[0]);
        bx[1] = o1 - (p1 + dp[1]);
        bx[2] = o2 - (p2 + dp[2]);
        bx[3] = o0 + (p3 + dp[3]);
        bx[4] = o1 + (p4 + dp[4]);
        bx[5] = o2 + (p5 + dp[5]);
        unsigned key = 0u;
        if (s > SCORE_TH) {
            const unsigned sb = __float_as_uint(s) >> 12;   // 19-bit monotone score
            key = (sb << 13) | (8191u - (unsigned)m);       // low bits: tiebreak, smaller m first
        }
        keys[mb] = key;
    }
}

// =====================================================================
// K2: per-batch descending bitonic sort of 8192 keys in LDS.
// Keys staged via global_load_async_to_lds (CDNA5 async copy path).
// Gather emits padded 32B records (b128 x2) + ballot-packed valid bits.
// grid = B_, block = 1024
// =====================================================================
__global__ void __launch_bounds__(1024)
nms_sort_kernel(const unsigned* __restrict__ keys,
                const float* __restrict__ scores, const float* __restrict__ boxes,
                const int* __restrict__ cls,
                float* __restrict__ srec, unsigned* __restrict__ vbits)
{
    const int b   = blockIdx.x;
    const int tid = threadIdx.x;
    __shared__ unsigned skey[SORTN];

    // async stage global keys -> LDS (one b32 per lane), zero-pad the tail
    for (int s = 0; s < 8; ++s) {
        const int i = tid + s * 1024;
        if (i < M_) {
            unsigned lds_addr = (unsigned)(uintptr_t)(&skey[i]);   // flat[31:0] == LDS byte addr
            unsigned long long ga =
                (unsigned long long)(uintptr_t)(keys + (size_t)b * M_ + i);
            asm volatile("global_load_async_to_lds_b32 %0, %1, off"
                         :: "v"(lds_addr), "v"(ga) : "memory");
        } else {
            skey[i] = 0u;
        }
    }
    asm volatile("s_wait_asynccnt 0x0" ::: "memory");
    __syncthreads();

    // bitonic, descending
    for (unsigned sz = 2; sz <= SORTN; sz <<= 1) {
        for (unsigned st = sz >> 1; st > 0; st >>= 1) {
            __syncthreads();
            for (int s = 0; s < 8; ++s) {
                const unsigned i = tid + s * 1024;
                const unsigned j = i ^ st;
                if (j > i) {
                    const unsigned a = skey[i], q = skey[j];
                    const bool up = ((i & sz) == 0);
                    if (up ? (a < q) : (a > q)) { skey[i] = q; skey[j] = a; }
                }
            }
        }
    }
    __syncthreads();

    // gather sorted candidate records: {box[6], score, class-bits}
    for (int s = 0; s < 5; ++s) {
        const int i = tid + s * 1024;
        if (i >= M_) continue;                 // wave-uniform (M_ % 32 == 0)
        const unsigned key = skey[i];
        const size_t ib = (size_t)b * M_ + i;
        const bool valid = key >= (1u << 13);

        float4 r0 = {0.f, 0.f, 0.f, 0.f};
        float4 r1 = {0.f, 0.f, 0.f, __int_as_float(-1)};
        if (valid) {
            const int m = 8191 - (int)(key & 8191u);
            const size_t mb = (size_t)b * M_ + m;
            r0.x = boxes[mb * 6 + 0]; r0.y = boxes[mb * 6 + 1];
            r0.z = boxes[mb * 6 + 2]; r0.w = boxes[mb * 6 + 3];
            r1.x = boxes[mb * 6 + 4]; r1.y = boxes[mb * 6 + 5];
            r1.z = scores[mb];
            r1.w = __int_as_float(cls[mb]);
        }
        float4* rp = (float4*)(srec + ib * 8);
        rp[0] = r0;
        rp[1] = r1;

        const unsigned long long bm = __ballot(valid);   // wave32: bits 0..31
        if ((tid & 31) == 0) vbits[(size_t)b * MW_ + (i >> 5)] = (unsigned)bm;
    }
}

// =====================================================================
// K3: suppression bitmask. mask[b][i][w] bit j%32 = (class match && IoU>thr).
// Classes must match (explicit class-id compare skips cross-class pairs).
// grid = B_ * (M_/32) * (MW_/8), block = 256. LDS-tiled 32 rows x 256 cols.
// =====================================================================
__global__ void __launch_bounds__(256)
nms_mask_kernel(const float* __restrict__ srec, unsigned* __restrict__ mask)
{
    const int bid  = blockIdx.x;
    const int wblk = bid % (MW_ / 8);            // 18 word-blocks (256 cols each)
    const int rblk = (bid / (MW_ / 8)) % (M_ / 32);
    const int b    = bid / ((MW_ / 8) * (M_ / 32));
    const int tid  = threadIdx.x;

    __shared__ float rb[32][8];
    __shared__ float cb[256][8];

    const int j = wblk * 256 + tid;
    {
        const float4* src = (const float4*)(srec + ((size_t)b * M_ + j) * 8);
        float4* dst = (float4*)&cb[tid][0];
        dst[0] = src[0];
        dst[1] = src[1];
    }
    if (tid < 32) {
        const int i = rblk * 32 + tid;
        const float4* src = (const float4*)(srec + ((size_t)b * M_ + i) * 8);
        float4* dst = (float4*)&rb[tid][0];
        dst[0] = src[0];
        dst[1] = src[1];
    }
    __syncthreads();

    const int il = tid & 31;
    const int wl = tid >> 5;                      // local word 0..7
    float bi[6];
    for (int q = 0; q < 6; ++q) bi[q] = rb[il][q];
    const int ci = __float_as_int(rb[il][7]);
    const float va = (bi[3] - bi[0]) * (bi[4] - bi[1]) * (bi[5] - bi[2]);

    unsigned bits = 0u;
    for (int bit = 0; bit < 32; ++bit) {
        const int jl = wl * 32 + bit;
        if (__float_as_int(cb[jl][7]) != ci) continue;
        const float* bj = cb[jl];
        const float x0 = fmaxf(bi[0], bj[0]);
        const float y0 = fmaxf(bi[1], bj[1]);
        const float z0 = fmaxf(bi[2], bj[2]);
        const float x1 = fminf(bi[3], bj[3]);
        const float y1 = fminf(bi[4], bj[4]);
        const float z1 = fminf(bi[5], bj[5]);
        const float inter = fmaxf(x1 - x0, 0.f) * fmaxf(y1 - y0, 0.f) * fmaxf(z1 - z0, 0.f);
        const float vb = (bj[3] - bj[0]) * (bj[4] - bj[1]) * (bj[5] - bj[2]);
        const float iou = inter / (va + vb - inter + 1e-8f);
        if (iou > NMS_TH) bits |= (1u << bit);
    }
    const int i = rblk * 32 + il;
    mask[((size_t)b * M_ + i) * MW_ + (size_t)(wblk * 8 + wl)] = bits;
}

// =====================================================================
// K4: barrier-free single-wave greedy scan per batch.
// Suppression state (144 bits x 32 lanes) and validity bitvector live in
// 5 registers/lane; per-candidate check is two ds_bpermute (__shfl).
// Row-OR (5 coalesced L2 loads/lane) only on kept rows. Distance-16
// global_prefetch_b8 row lookahead hides L2 latency.
// grid = B_, block = 32 (one wave)
// =====================================================================
__global__ void __launch_bounds__(32)
nms_scan_kernel(const float* __restrict__ srec, const unsigned* __restrict__ vbits,
                const unsigned* __restrict__ mask, float* __restrict__ out)
{
    const int b    = blockIdx.x;
    const int lane = threadIdx.x;

    float* ob = out + (size_t)b * TOPK_ * 6;
    float* os = out + (size_t)B_ * TOPK_ * 6 + (size_t)b * TOPK_;
    float* oc = out + (size_t)B_ * TOPK_ * 6 + (size_t)B_ * TOPK_ + (size_t)b * TOPK_;

    // zero / -1 fill outputs
    for (int q = lane; q < TOPK_ * 6; q += 32) ob[q] = 0.f;
    for (int q = lane; q < TOPK_; q += 32) { os[q] = 0.f; oc[q] = -1.0f; }

    unsigned supp[5] = {0u, 0u, 0u, 0u, 0u};
    unsigned vb[5];
    for (int s = 0; s < 5; ++s) {
        const int w = lane + 32 * s;
        vb[s] = (w < MW_) ? vbits[(size_t)b * MW_ + w] : 0u;
    }

    int cnt = 0;
    const unsigned* mbase = mask + (size_t)b * M_ * MW_;

    // i>>10 selects the register slot; segment the loop so the index is constant
    for (int seg = 0; seg < 5; ++seg) {
        const int i0 = seg * 1024;
        const int i1 = (i0 + 1024 < M_) ? i0 + 1024 : M_;
        for (int i = i0; i < i1; ++i) {
            // speculative row lookahead (one 128B line per lane 0..4)
            const int ip = i + 16;
            if (ip < M_ && lane < 5)
                __builtin_prefetch((const char*)(mbase + (size_t)ip * MW_) + lane * 128, 0, 0);

            const int wlane = (i >> 5) & 31;
            const unsigned sw = __shfl(supp[seg], wlane);
            const unsigned vw = __shfl(vb[seg], wlane);
            const unsigned bit = 1u << (i & 31);
            const bool keep = (vw & bit) && !(sw & bit);   // wave-uniform

            if (keep) {
                const unsigned* mrow = mbase + (size_t)i * MW_;
                for (int s = 0; s < 5; ++s) {
                    const int w = lane + 32 * s;
                    if (w < MW_) supp[s] |= mrow[w];
                }
                if (cnt < TOPK_) {
                    const size_t ib = (size_t)b * M_ + i;
                    const float v = (lane < 8) ? srec[ib * 8 + lane] : 0.f;
                    const int pidx = (lane < 3) ? lane + 3 : ((lane < 6) ? lane - 3 : lane);
                    const float pv = __shfl(v, pidx);
                    if (lane < 3)       ob[cnt * 6 + lane] = 0.5f * (v + pv);   // center
                    else if (lane < 6)  ob[cnt * 6 + lane] = v - pv;            // size
                    else if (lane == 6) os[cnt] = v;                            // score
                    else if (lane == 7) oc[cnt] = (float)__float_as_int(v);     // class
                }
                ++cnt;
            }
        }
    }
}

extern "C" void kernel_launch(void* const* d_in, const int* in_sizes, int n_in,
                              void* d_out, int out_size, void* d_ws, size_t ws_size,
                              hipStream_t stream) {
    (void)in_sizes; (void)n_in; (void)out_size; (void)ws_size;
    const float* logits = (const float*)d_in[0];   // [B,N,19]
    const float* deltas = (const float*)d_in[1];   // [B,N,C,6]
    const float* props  = (const float*)d_in[2];   // [B,N,6]
    const float* origs  = (const float*)d_in[3];   // [B,N,3]
    float* out = (float*)d_out;

    char* ws = (char*)d_ws;
    float*    scores  = (float*)(ws + OFF_SCORES);
    float*    boxes   = (float*)(ws + OFF_BOXES);
    int*      cls     = (int*)(ws + OFF_CLS);
    unsigned* keys    = (unsigned*)(ws + OFF_KEYS);
    float*    srec    = (float*)(ws + OFF_SREC);
    unsigned* vbits   = (unsigned*)(ws + OFF_VBITS);
    unsigned* maskbuf = (unsigned*)(ws + OFF_MASK);

    nms_decode_kernel<<<dim3(B_ * 2), dim3(256), 0, stream>>>(
        logits, deltas, props, origs, scores, boxes, cls, keys);

    nms_sort_kernel<<<dim3(B_), dim3(1024), 0, stream>>>(
        keys, scores, boxes, cls, srec, vbits);

    nms_mask_kernel<<<dim3(B_ * (M_ / 32) * (MW_ / 8)), dim3(256), 0, stream>>>(
        srec, maskbuf);

    nms_scan_kernel<<<dim3(B_), dim3(32), 0, stream>>>(
        srec, vbits, maskbuf, out);
}
